// YoloxLoss_18897856103100
// MI455X (gfx1250) — matrix operand order repaired and checked by reading, not compiled, not run on hardware
//
#include <hip/hip_runtime.h>
#include <math.h>

typedef __attribute__((ext_vector_type(16))) _Float16 v16h;
typedef __attribute__((ext_vector_type(8)))  float    v8f;

#define CC 80          // num classes
#define CP 96          // classes padded to 3 WMMA k-chunks of 32
#define MM 50          // max GT boxes
#define BIGV 1000000000.0f

__device__ __forceinline__ float sigm(float x) { return 1.0f / (1.0f + __expf(-x)); }
__device__ __forceinline__ float bce(float x, float t) {
    return fmaxf(x, 0.0f) - x * t + log1pf(__expf(-fabsf(x)));
}

// ---------------------------------------------------------------------------
// K1: per (b,a) anchor stats: Y[a,c] = logp - log1mp (f16, padded), S = sum log1mp,
//     fg = any_m(in_box | in_ctr)
// ---------------------------------------------------------------------------
__global__ void yolox_k1_anchor(const float* __restrict__ outputs,
                                const float* __restrict__ xs, const float* __restrict__ ys,
                                const float* __restrict__ es, const float* __restrict__ labels,
                                _Float16* __restrict__ Y, float* __restrict__ S,
                                int* __restrict__ FG, int A)
{
    int a = blockIdx.x * blockDim.x + threadIdx.x;
    int b = blockIdx.y;
    if (a >= A) return;
    const float* o = outputs + ((size_t)b * A + a) * (5 + CC);
    float so = sigm(o[4]);
    float s_sum = 0.0f;
    _Float16* y = Y + ((size_t)b * A + a) * CP;
    #pragma unroll 4
    for (int c = 0; c < CP; ++c) {
        float v = 0.0f;
        if (c < CC) {
            float p  = sqrtf(sigm(o[5 + c]) * so);
            float lp = fmaxf(__logf(p), -100.0f);
            float l1 = fmaxf(log1pf(-p), -100.0f);
            s_sum += l1;
            v = lp - l1;
        }
        y[c] = (_Float16)v;
    }
    S[(size_t)b * A + a] = s_sum;

    float st  = es[a];
    float xc  = (xs[a] + 0.5f) * st;
    float yc  = (ys[a] + 0.5f) * st;
    float rad = 2.5f * st;
    int fg = 0;
    const float* lab = labels + (size_t)b * MM * 5;
    for (int m = 0; m < MM; ++m) {
        const float* g = lab + m * 5;
        float gx = g[0], gy = g[1], gw = g[2], gh = g[3];
        if (!((gx + gy + gw + gh + g[4]) > 0.0f)) continue;
        bool inb = (xc > gx - 0.5f * gw) && (gx + 0.5f * gw > xc) &&
                   (yc > gy - 0.5f * gh) && (gy + 0.5f * gh > yc);
        bool inc = (xc > gx - rad) && (xc < gx + rad) && (yc > gy - rad) && (yc < gy + rad);
        if (inb | inc) { fg = 1; break; }
    }
    FG[(size_t)b * A + a] = fg;
}

// ---------------------------------------------------------------------------
// K1b: build the one-hot A-operand PRE-SWIZZLED into WMMA A-matrix VGPR layout.
//   OHsw[((b*4 + mt)*3 + ch)*32 + lane][0..15] : lane row = lane%16,
//   element i -> K = ch*32 + (lane>>4)*16 + i.
// 12 KB per batch, reused by every anchor block of that batch (L2-hot).
// ---------------------------------------------------------------------------
__global__ void yolox_k1b_onehot(const float* __restrict__ labels,
                                 _Float16* __restrict__ OHsw, int total)
{
    int t = blockIdx.x * blockDim.x + threadIdx.x;
    if (t >= total) return;
    int lane = t & 31;
    int ch   = (t >> 5) % 3;
    int mt   = (t / 96) & 3;
    int b    = t / 384;
    int mrow = mt * 16 + (lane & 15);
    int hi   = lane >> 4;
    int gtc  = (mrow < MM) ? (int)labels[((size_t)b * MM + mrow) * 5 + 4] : -1;
    v16h h;
    #pragma unroll
    for (int i = 0; i < 16; ++i) {
        int k = ch * 32 + hi * 16 + i;
        h[i] = (_Float16)((k == gtc) ? 1.0f : 0.0f);
    }
    *(v16h*)(OHsw + (size_t)t * 16) = h;
}

// ---------------------------------------------------------------------------
// K2: pair cost + IoU. cls term via V_WMMA_F32_16X16X32_F16:
//   D(16 gt x 16 anchor) = OneHot(16x32 f16, preswizzled load) x Y-chunk(32x16)
// Block = 128 threads (4 waves); wave w owns anchors [blk*64 + w*16, +16),
// loops 4 GT row-tiles (M padded to 64) x 3 K-chunks (C padded to 96).
// ---------------------------------------------------------------------------
__global__ void yolox_k2_cost(const float* __restrict__ outputs,
                              const float* __restrict__ xs, const float* __restrict__ ys,
                              const float* __restrict__ es, const float* __restrict__ labels,
                              const _Float16* __restrict__ Y, const _Float16* __restrict__ OHsw,
                              const float* __restrict__ S, const int* __restrict__ FG,
                              float* __restrict__ cost, float* __restrict__ iouP, int A)
{
    __shared__ float slab[MM * 5];
    int b = blockIdx.y;
    for (int i = threadIdx.x; i < MM * 5; i += blockDim.x)
        slab[i] = labels[(size_t)b * MM * 5 + i];
    __syncthreads();

    int wave = threadIdx.x >> 5;
    int lane = threadIdx.x & 31;
    int al   = lane & 15;        // anchor column within tile
    int hi   = lane >> 4;        // lane half (K/M split per ISA layout)
    int a0   = blockIdx.x * 64 + wave * 16;
    int a    = a0 + al; if (a > A - 1) a = A - 1;   // clamp: EXEC stays all-ones

    const float* o = outputs + ((size_t)b * A + a) * (5 + CC);
    float px = o[0], py = o[1], pw = o[2], ph = o[3];
    float Sa  = S[(size_t)b * A + a];
    int   fga = FG[(size_t)b * A + a];
    float st  = es[a];
    float xc  = (xs[a] + 0.5f) * st;
    float yc  = (ys[a] + 0.5f) * st;
    float rad = 2.5f * st;

    // B operand: lane holds column a, 16 contiguous K values per half-wave
    const _Float16* yrow = Y + ((size_t)b * A + a) * CP + hi * 16;
    v16h bm0 = *(const v16h*)(yrow);
    v16h bm1 = *(const v16h*)(yrow + 32);
    v16h bm2 = *(const v16h*)(yrow + 64);
    // prefetch next anchor block's Y rows (global_prefetch_b8)
    __builtin_prefetch((const char*)(yrow + (size_t)64 * CP), 0, 3);

    const _Float16* ohb = OHsw + (size_t)b * 4 * 3 * 32 * 16;

    for (int mt = 0; mt < 4; ++mt) {
        v8f acc = {};
        #pragma unroll
        for (int ch = 0; ch < 3; ++ch) {
            v16h am = *(const v16h*)(ohb + (((size_t)mt * 3 + ch) * 32 + lane) * 16);
            v16h bm = (ch == 0) ? bm0 : ((ch == 1) ? bm1 : bm2);
            acc = __builtin_amdgcn_wmma_f32_16x16x32_f16(
                      false, am, false, bm, (short)0, acc, false, false);
        }
        // C/D layout: VGPR r, lane<16 -> M=r ; lane>=16 -> M=r+8 ; N=lane%16
        #pragma unroll
        for (int r = 0; r < 8; ++r) {
            int m  = mt * 16 + r + hi * 8;
            int aa = a0 + al;
            if (m < MM && aa < A) {
                const float* g = slab + m * 5;
                float gx = g[0], gy = g[1], gw = g[2], gh = g[3];
                bool valid = (gx + gy + gw + gh + g[4]) > 0.0f;
                float tlx = fmaxf(gx - 0.5f * gw, px - 0.5f * pw);
                float tly = fmaxf(gy - 0.5f * gh, py - 0.5f * ph);
                float brx = fminf(gx + 0.5f * gw, px + 0.5f * pw);
                float bry = fminf(gy + 0.5f * gh, py + 0.5f * ph);
                bool  en  = (tlx < brx) && (tly < bry);
                float ai  = (brx - tlx) * (bry - tly) * (en ? 1.0f : 0.0f);
                float iou = ai / (gw * gh + pw * ph - ai + 1e-16f);
                iou = (fga && valid) ? iou : 0.0f;
                bool inb = (xc > gx - 0.5f * gw) && (gx + 0.5f * gw > xc) &&
                           (yc > gy - 0.5f * gh) && (gy + 0.5f * gh > yc);
                bool inc = (xc > gx - rad) && (xc < gx + rad) &&
                           (yc > gy - rad) && (yc < gy + rad);
                bool inboth = inb && inc && valid;
                float cls_cost = -(Sa + acc[r]);
                float cst = cls_cost - 3.0f * __logf(iou + 1e-8f) + (inboth ? 0.0f : 100000.0f);
                cst = (fga && valid) ? cst : BIGV;
                size_t idx = ((size_t)b * MM + m) * A + aa;
                cost[idx] = cst;
                iouP[idx] = iou;
            }
        }
    }
}

// ---------------------------------------------------------------------------
// K3: dyn_k = max(1, int(sum(top10 iou))) per (b,m). One wave per GT row.
// ---------------------------------------------------------------------------
__global__ void yolox_k3_dynk(const float* __restrict__ iouP, int* __restrict__ dynk, int A)
{
    int m = blockIdx.x, b = blockIdx.y;
    int lane = threadIdx.x;
    const float* row = iouP + ((size_t)b * MM + m) * A;
    float t[10];
    #pragma unroll
    for (int i = 0; i < 10; ++i) t[i] = 0.0f;
    for (int a = lane; a < A; a += 32) {
        float v = row[a];
        if (v > t[9]) {
            t[9] = v;
            #pragma unroll
            for (int i = 9; i > 0; --i) {
                if (t[i] > t[i - 1]) { float tp = t[i - 1]; t[i - 1] = t[i]; t[i] = tp; }
            }
        }
    }
    __shared__ float buf[320];
    #pragma unroll
    for (int i = 0; i < 10; ++i) buf[lane * 10 + i] = t[i];
    __syncthreads();
    if (lane == 0) {
        float s[10];
        #pragma unroll
        for (int i = 0; i < 10; ++i) s[i] = 0.0f;
        for (int i = 0; i < 320; ++i) {
            float v = buf[i];
            if (v > s[9]) {
                s[9] = v;
                #pragma unroll
                for (int j = 9; j > 0; --j) {
                    if (s[j] > s[j - 1]) { float tp = s[j - 1]; s[j - 1] = s[j]; s[j] = tp; }
                }
            }
        }
        float sum = 0.0f;
        #pragma unroll
        for (int i = 0; i < 10; ++i) sum += s[i];
        int k = (int)sum; if (k < 1) k = 1; if (k > 10) k = 10;
        dynk[b * MM + m] = k;
    }
}

// ---------------------------------------------------------------------------
// K4: select dyn_k lowest-cost anchors per (b,m) (ties -> lowest index).
// ---------------------------------------------------------------------------
__global__ void yolox_k4_match(const float* __restrict__ cost, const int* __restrict__ dynk,
                               unsigned char* __restrict__ match, int A)
{
    int m = blockIdx.x, b = blockIdx.y;
    int lane = threadIdx.x;
    size_t base = ((size_t)b * MM + m) * A;
    unsigned char* mr = match + base;
    const float* cr = cost + base;
    for (int a = lane; a < A; a += 32) mr[a] = 0;
    __syncthreads();
    int k = dynk[b * MM + m];
    __shared__ float vbuf[32];
    __shared__ int ibuf[32];
    __shared__ int chosen;
    for (int it = 0; it < k; ++it) {
        float bv = BIGV; int bi = A;
        for (int a = lane; a < A; a += 32) {
            if (mr[a]) continue;
            float v = cr[a];
            if (v < bv) { bv = v; bi = a; }
        }
        vbuf[lane] = bv; ibuf[lane] = bi;
        __syncthreads();
        if (lane == 0) {
            float mv = vbuf[0]; int mi = ibuf[0];
            for (int l = 1; l < 32; ++l) {
                if (vbuf[l] < mv || (vbuf[l] == mv && ibuf[l] < mi)) { mv = vbuf[l]; mi = ibuf[l]; }
            }
            if (mv < BIGV * 0.5f && mi < A) { mr[mi] = 1; chosen = 1; }
            else chosen = 0;
        }
        __syncthreads();
        if (!chosen) break;
    }
}

// ---------------------------------------------------------------------------
// K5: per (b,a): resolve multi-match by argmin cost, then loss terms.
// ---------------------------------------------------------------------------
__global__ void yolox_k5_loss(const float* __restrict__ outputs, const float* __restrict__ labels,
                              const float* __restrict__ cost, const float* __restrict__ iouP,
                              const unsigned char* __restrict__ match,
                              float* __restrict__ partials, int A, int nblkA, int nblk)
{
    int a = blockIdx.x * 256 + threadIdx.x;
    int b = blockIdx.y;
    float t_iou = 0.0f, t_obj = 0.0f, t_cls = 0.0f, t_fg = 0.0f;
    if (a < A) {
        int amg = 0, first = 0;
        for (int m = 0; m < MM; ++m) {
            if (match[((size_t)b * MM + m) * A + a]) { if (amg == 0) first = m; amg++; }
        }
        int mgt = 0; bool fgf = false;
        if (amg > 1) {
            float mv = 3.0e9f; int mi = 0;
            for (int m = 0; m < MM; ++m) {
                float v = cost[((size_t)b * MM + m) * A + a];
                if (v < mv) { mv = v; mi = m; }
            }
            mgt = mi; fgf = true;
        } else if (amg == 1) { mgt = first; fgf = true; }
        const float* o = outputs + ((size_t)b * A + a) * (5 + CC);
        t_obj = bce(o[4], fgf ? 1.0f : 0.0f);
        if (fgf) {
            t_fg = 1.0f;
            float pious = iouP[((size_t)b * MM + mgt) * A + a];
            const float* g = labels + ((size_t)b * MM + mgt) * 5;
            int gtc = (int)g[4];
            float px = o[0], py = o[1], pw = o[2], ph = o[3];
            float gx = g[0], gy = g[1], gw = g[2], gh = g[3];
            float tlx = fmaxf(gx - 0.5f * gw, px - 0.5f * pw);
            float tly = fmaxf(gy - 0.5f * gh, py - 0.5f * ph);
            float brx = fminf(gx + 0.5f * gw, px + 0.5f * pw);
            float bry = fminf(gy + 0.5f * gh, py + 0.5f * ph);
            bool  en  = (tlx < brx) && (tly < bry);
            float ai  = (brx - tlx) * (bry - tly) * (en ? 1.0f : 0.0f);
            float iou = ai / (pw * ph + gw * gh - ai + 1e-16f);
            t_iou = 1.0f - iou * iou;
            for (int c = 0; c < CC; ++c)
                t_cls += bce(o[5 + c], (c == gtc) ? pious : 0.0f);
        }
    }
    __shared__ float red[256];
    int tid = threadIdx.x;
    int blk = b * nblkA + blockIdx.x;
    float vals[4] = { t_iou, t_obj, t_cls, t_fg };
    for (int q = 0; q < 4; ++q) {
        red[tid] = vals[q]; __syncthreads();
        for (int s = 128; s > 0; s >>= 1) { if (tid < s) red[tid] += red[tid + s]; __syncthreads(); }
        if (tid == 0) partials[(size_t)q * nblk + blk] = red[0];
        __syncthreads();
    }
}

// ---------------------------------------------------------------------------
// K6: deterministic fixed-order final reduction -> scalar loss.
// ---------------------------------------------------------------------------
__global__ void yolox_k6_final(const float* __restrict__ partials, int nblk, float* __restrict__ out)
{
    __shared__ float red[256];
    __shared__ float tot[4];
    int tid = threadIdx.x;
    for (int q = 0; q < 4; ++q) {
        float s = 0.0f;
        for (int i = tid; i < nblk; i += 256) s += partials[(size_t)q * nblk + i];
        red[tid] = s; __syncthreads();
        for (int st = 128; st > 0; st >>= 1) { if (tid < st) red[tid] += red[tid + st]; __syncthreads(); }
        if (tid == 0) tot[q] = red[0];
        __syncthreads();
    }
    if (tid == 0) {
        float nf = fmaxf(tot[3], 1.0f);
        out[0] = (5.0f * tot[0] + tot[1] + tot[2]) / nf;
    }
}

extern "C" void kernel_launch(void* const* d_in, const int* in_sizes, int n_in,
                              void* d_out, int out_size, void* d_ws, size_t ws_size,
                              hipStream_t stream)
{
    const float* outputs = (const float*)d_in[0];
    const float* xs      = (const float*)d_in[1];
    const float* ys      = (const float*)d_in[2];
    const float* es      = (const float*)d_in[3];
    const float* labels  = (const float*)d_in[4];

    int A = in_sizes[1];                       // 8400
    int B = in_sizes[0] / (A * (5 + CC));      // 32

    // workspace carve-out (256B aligned slices)
    char* p = (char*)d_ws;
    auto carve = [&](size_t bytes) -> char* {
        char* r = p; p += (bytes + 255) & ~(size_t)255; return r;
    };
    _Float16*      Y      = (_Float16*)     carve((size_t)B * A * CP * sizeof(_Float16));
    _Float16*      OHsw   = (_Float16*)     carve((size_t)B * 4 * 3 * 32 * 16 * sizeof(_Float16));
    float*         S      = (float*)        carve((size_t)B * A * sizeof(float));
    int*           FG     = (int*)          carve((size_t)B * A * sizeof(int));
    float*         cost   = (float*)        carve((size_t)B * MM * A * sizeof(float));
    float*         iouP   = (float*)        carve((size_t)B * MM * A * sizeof(float));
    int*           dynk   = (int*)          carve((size_t)B * MM * sizeof(int));
    unsigned char* match  = (unsigned char*)carve((size_t)B * MM * A);
    int nblkA = (A + 255) / 256;
    int nblk  = nblkA * B;
    float*         parts  = (float*)        carve((size_t)4 * nblk * sizeof(float));

    int ohTotal = B * 4 * 3 * 32;
    yolox_k1_anchor<<<dim3((A + 255) / 256, B), 256, 0, stream>>>(
        outputs, xs, ys, es, labels, Y, S, FG, A);
    yolox_k1b_onehot<<<(ohTotal + 255) / 256, 256, 0, stream>>>(labels, OHsw, ohTotal);
    yolox_k2_cost<<<dim3((A + 63) / 64, B), 128, 0, stream>>>(
        outputs, xs, ys, es, labels, Y, OHsw, S, FG, cost, iouP, A);
    yolox_k3_dynk<<<dim3(MM, B), 32, 0, stream>>>(iouP, dynk, A);
    yolox_k4_match<<<dim3(MM, B), 32, 0, stream>>>(cost, dynk, match, A);
    yolox_k5_loss<<<dim3(nblkA, B), 256, 0, stream>>>(
        outputs, labels, cost, iouP, match, parts, A, nblkA, nblk);
    yolox_k6_final<<<1, 256, 0, stream>>>(parts, nblk, (float*)d_out);
}
